// ICLMModel_9929964388548
// MI455X (gfx1250) — compile-verified, matrix operand order checked by reading, not run on hardware
//
#include <hip/hip_runtime.h>
#include <hip/hip_bf16.h>

// ICLM t>=1 path for MI455X (gfx1250, wave32, WMMA bf16 + TDM async tensor load)
//
// out[m,n,l] = hidden[m,l] * sum_r w_probs[l,r] * sum_k A[m*25+r,k] * P[k, n*16+l]
// hidden[m,l] = gate( clamp01( clamp01(rowsum(A rows of m)) @ h_probs^T ) )

#define N_ENT 1200
#define R_REL 25             // n+1
#define LCH   16             // rule channels
#define NM    512            // M*L columns of the GEMM
#define KDIM  1200
#define KPAD  1216           // KDIM padded to multiple of 32/64 (zero filled)
#define KC    32             // bf16 WMMA K per step
#define KTILES (KPAD / KC)   // 38

#define WS_BT_BYTE_OFF 4096  // Bt (bf16 [512][1216]) starts here in d_ws

typedef __attribute__((ext_vector_type(16))) __bf16        v16bf;
typedef __attribute__((ext_vector_type(8)))  float         v8f;
typedef __attribute__((ext_vector_type(4)))  unsigned int  v4u;
typedef __attribute__((ext_vector_type(8)))  int           v8i;
typedef __attribute__((ext_vector_type(4)))  int           v4i;

union V16U { uint4 u[2]; v16bf v; };

__device__ __forceinline__ float clamp01f(float x) { return fminf(fmaxf(x, 0.0f), 1.0f); }

__device__ __forceinline__ unsigned int lds_off32(const void* p) {
    // generic LDS pointer: low 32 bits are the in-LDS byte offset
    return (unsigned int)(uintptr_t)p;
}

// ---------------------------------------------------------------------------
// TDM: async 2D tile load Bt[tile: 512 lines x 32 bf16] -> LDS (contiguous 32KB)
//   tensor: dim0 = KPAD elements (2B), dim1 = 512 lines, line stride = KPAD
// Issued by one wave; completion via TENSORcnt + workgroup barrier.
// ---------------------------------------------------------------------------
__device__ __forceinline__ void tdm_load_B(const __hip_bfloat16* tile_src,
                                           unsigned int lds_byte_off) {
    const unsigned long long ga = (unsigned long long)(uintptr_t)tile_src;
    v4u g0;
    g0[0] = 1u;                                   // count=1 (valid), is_restore=0
    g0[1] = lds_byte_off;                         // bits 63:32  lds_addr
    g0[2] = (unsigned int)ga;                     // bits 95:64  global_addr[31:0]
    g0[3] = (unsigned int)((ga >> 32) & 0x01FFFFFFu) | (2u << 30); // addr[56:32] | type=2
    v8i g1;
    g1[0] = (int)(1u << 16);                      // workgroup_mask=0, data_size=1 (2B)
    g1[1] = (int)((KPAD & 0xFFFFu) << 16);        // tensor_dim0[15:0] @ bits 63:48
    g1[2] = (int)((KPAD >> 16) | (512u << 16));   // tensor_dim0[31:16] | tensor_dim1[15:0]
    g1[3] = (int)((512u >> 16) | ((unsigned)KC << 16)); // tensor_dim1[31:16] | tile_dim0=32
    g1[4] = (int)512;                             // tile_dim1=512, tile_dim2=0
    g1[5] = (int)KPAD;                            // tensor_dim0_stride[31:0]
    g1[6] = 0;                                    // stride hi | dim1_stride lo
    g1[7] = 0;
    v4i gz = {0, 0, 0, 0};
#if __clang_major__ >= 23
    v8i gz8 = {0, 0, 0, 0, 0, 0, 0, 0};
    __builtin_amdgcn_tensor_load_to_lds(g0, g1, gz, gz, gz8, 0);
#else
    __builtin_amdgcn_tensor_load_to_lds(g0, g1, gz, gz, 0);
#endif
}

// ---------------------------------------------------------------------------
// Prep: w_probs = softmax(w[t]) [16x25] -> ws[0..399]
//       hp      = clamp01(h[t]/10) cols 0..24, col 24 forced 0 -> ws[400..799]
//       p       = clamp01(h[t][:,25]/10) -> ws[800..815]
// ---------------------------------------------------------------------------
__global__ void iclm_prep(const float* __restrict__ w,
                          const float* __restrict__ h,
                          const int*   __restrict__ t_ptr,
                          float* __restrict__ ws) {
    const int t = t_ptr[0];
    const int l = threadIdx.x;
    if (l >= LCH) return;
    const float* wt = w + (size_t)t * LCH * R_REL       + (size_t)l * R_REL;
    const float* ht = h + (size_t)t * LCH * (R_REL + 1) + (size_t)l * (R_REL + 1);

    float mx = -3.4e38f;
    for (int r = 0; r < R_REL; ++r) mx = fmaxf(mx, wt[r]);
    float s = 0.0f;
    for (int r = 0; r < R_REL; ++r) s += __expf(wt[r] - mx);
    const float inv = 1.0f / s;
    for (int r = 0; r < R_REL; ++r) ws[l * R_REL + r] = __expf(wt[r] - mx) * inv;

    for (int r = 0; r < R_REL; ++r) {
        float v = clamp01f(ht[r] * 0.1f);
        if (r == R_REL - 1) v = 0.0f;            // h_probs[:, -2] := 0 (index 24)
        ws[400 + l * R_REL + r] = v;
    }
    ws[800 + l] = clamp01f(ht[R_REL] * 0.1f);    // p = h_probs[:, -1]
}

// ---------------------------------------------------------------------------
// One-time transpose + convert: Bt[c][k] = bf16(P[k][c]), k padded to 1216 w/ 0.
// 64x64 tiles through LDS; grid = 8 col-tiles x 19 k-tiles.
// ---------------------------------------------------------------------------
__global__ __launch_bounds__(256)
void iclm_cvtB(const float* __restrict__ P, __hip_bfloat16* __restrict__ Bt) {
    __shared__ float tile[64][65];
    const int kt = blockIdx.x % (KPAD / 64);     // 19
    const int ct = blockIdx.x / (KPAD / 64);     // 8
    const int k0 = kt * 64, c0 = ct * 64;
    const int tx = threadIdx.x & 63;
    const int ty = threadIdx.x >> 6;             // 0..3
    #pragma unroll
    for (int i = 0; i < 16; ++i) {
        const int k = k0 + ty + i * 4;
        tile[ty + i * 4][tx] = (k < KDIM) ? P[(size_t)k * NM + c0 + tx] : 0.0f;
    }
    __syncthreads();
    #pragma unroll
    for (int i = 0; i < 16; ++i) {
        const int c = c0 + ty + i * 4;
        const int k = k0 + tx;
        Bt[(size_t)c * KPAD + k] = __float2bfloat16(tile[tx][ty + i * 4]);
    }
}

// ---------------------------------------------------------------------------
// Main: one workgroup per entity m. 8 waves (256 threads).
// Wave wid: m_tile = wid&1 (rows 0..15 / 16..31 of padded 32-row A block),
//           col strip cs = wid>>1 (128 columns = 8 WMMA N-tiles).
// Double-buffered: TDM B(k+1) + global A(k+1) staged while computing k.
// ---------------------------------------------------------------------------
__global__ __launch_bounds__(256)
void iclm_main(const float* __restrict__ A,              // input_all [30000,1200]
               const __hip_bfloat16* __restrict__ Bt,    // [512][1216] bf16
               const float* __restrict__ ws,             // wp/hp/p
               float* __restrict__ out) {                // [1200,512]
    __shared__ __hip_bfloat16 As[2][32 * KC];   // [row][kk]
    __shared__ __hip_bfloat16 Bs[2][NM * KC];   // [col][kk] — TDM destination
    __shared__ float wp_s[LCH * R_REL];
    __shared__ float hp_s[LCH * R_REL];
    __shared__ float p_s[LCH];
    __shared__ float rowsum_s[32];
    __shared__ float hidden_s[LCH];
    __shared__ float out_s[NM];

    const int tid  = threadIdx.x;
    const int lane = tid & 31;
    const int wid  = tid >> 5;
    const int mt   = wid & 1;
    const int cs   = wid >> 1;
    const int nloc = lane & 15;
    const int koff = (lane < 16) ? 0 : 8;       // ISA 16-bit A/B lane layout
    const int m    = blockIdx.x;

    for (int i = tid; i < LCH * R_REL; i += 256) { wp_s[i] = ws[i]; hp_s[i] = ws[400 + i]; }
    if (tid < LCH) p_s[tid] = ws[800 + tid];
    if (tid < 32)  rowsum_s[tid] = 0.0f;

    v8f acc[8];
    #pragma unroll
    for (int i = 0; i < 8; ++i)
        #pragma unroll
        for (int j = 0; j < 8; ++j) acc[i][j] = 0.0f;

    const int arow = tid >> 3;                  // 0..31
    const int akb  = (tid & 7) * 4;             // 0,4,...,28
    const float* Aent = A + (size_t)m * R_REL * KDIM;

    // ---- stage A(fp32->bf16, fused rowsum) for one chunk ----
    auto stage_A = [&](int k0, __hip_bfloat16* dst) {
        float4 f = make_float4(0.f, 0.f, 0.f, 0.f);
        if (arow < R_REL && (k0 + akb) < KDIM)
            f = *(const float4*)(Aent + (size_t)arow * KDIM + k0 + akb);
        __hip_bfloat16* d = dst + arow * KC + akb;
        d[0] = __float2bfloat16(f.x);
        d[1] = __float2bfloat16(f.y);
        d[2] = __float2bfloat16(f.z);
        d[3] = __float2bfloat16(f.w);
        const float s = f.x + f.y + f.z + f.w;
        if (s != 0.0f) atomicAdd(&rowsum_s[arow], s);  // ~0.5% dense: mostly skipped
    };

    // ---- prologue: fill buffer 0 ----
    stage_A(0, As[0]);
    if (wid == 0) {
        tdm_load_B(Bt, lds_off32(&Bs[0][0]));
        __builtin_amdgcn_s_wait_tensorcnt(0);
    }
    __syncthreads();

    for (int kc = 0; kc < KTILES; ++kc) {
        const int buf = kc & 1;

        // prefetch next chunk into the other buffer (overlaps with WMMAs below)
        if (kc + 1 < KTILES) {
            stage_A((kc + 1) * KC, As[buf ^ 1]);
            if (wid == 0)
                tdm_load_B(Bt + (size_t)(kc + 1) * KC, lds_off32(&Bs[buf ^ 1][0]));
        }

        // ---- WMMA: 1 A-frag, 8 N-tiles per wave ----
        V16U a;
        const __hip_bfloat16* ap = &As[buf][(mt * 16 + nloc) * KC + koff];
        a.u[0] = *(const uint4*)(ap);        // K 0..7   (hi lanes: 8..15)
        a.u[1] = *(const uint4*)(ap + 16);   // K 16..23 (hi lanes: 24..31)
        #pragma unroll
        for (int nt = 0; nt < 8; ++nt) {
            V16U b;
            const __hip_bfloat16* bp = &Bs[buf][(cs * 128 + nt * 16 + nloc) * KC + koff];
            b.u[0] = *(const uint4*)(bp);
            b.u[1] = *(const uint4*)(bp + 16);
            acc[nt] = __builtin_amdgcn_wmma_f32_16x16x32_bf16(
                false, a.v, false, b.v, (short)0, acc[nt], false, false);
        }

        if (wid == 0) __builtin_amdgcn_s_wait_tensorcnt(0);  // next B tile landed
        __syncthreads();
    }

    // ---- epilogue: hidden gate + relation contraction, all in LDS ----
    out_s[tid] = 0.0f;
    out_s[tid + 256] = 0.0f;
    if (tid < LCH) {
        float s = 0.0f;
        #pragma unroll
        for (int r = 0; r < R_REL; ++r)
            s += clamp01f(rowsum_s[r]) * hp_s[tid * R_REL + r];
        const float hdn = clamp01f(s);
        const float p = p_s[tid];
        hidden_s[tid] = hdn * (1.0f - p) + p;
    }
    __syncthreads();

    // D layout: VGPR j holds row (mt*16 + koff + j); lane&15 selects column.
    #pragma unroll
    for (int nt = 0; nt < 8; ++nt) {
        const int c = cs * 128 + nt * 16 + nloc;   // lch = c&15 = nloc
        float partial = 0.0f;
        #pragma unroll
        for (int j = 0; j < 8; ++j) {
            const int r = mt * 16 + koff + j;      // relation (rows 25..31 are padding)
            if (r < R_REL) partial += acc[nt][j] * wp_s[nloc * R_REL + r];
        }
        atomicAdd(&out_s[c], partial);
    }
    __syncthreads();

    float* outm = out + (size_t)m * NM;
    outm[tid]       = out_s[tid]       * hidden_s[tid & 15];
    outm[tid + 256] = out_s[tid + 256] * hidden_s[tid & 15];
}

// ---------------------------------------------------------------------------
extern "C" void kernel_launch(void* const* d_in, const int* in_sizes, int n_in,
                              void* d_out, int out_size, void* d_ws, size_t ws_size,
                              hipStream_t stream) {
    // setup_inputs order: inp(0, unused), input_all(1), prev_state(2),
    //                     w(3), h(4), h_y(5, unused), t(6)
    const float* input_all  = (const float*)d_in[1];
    const float* prev_state = (const float*)d_in[2];
    const float* w          = (const float*)d_in[3];
    const float* h          = (const float*)d_in[4];
    const int*   t          = (const int*)d_in[6];
    float* ws = (float*)d_ws;
    __hip_bfloat16* Bt = (__hip_bfloat16*)((char*)d_ws + WS_BT_BYTE_OFF);
    float* out = (float*)d_out;

    hipLaunchKernelGGL(iclm_prep, dim3(1), dim3(32), 0, stream, w, h, t, ws);
    hipLaunchKernelGGL(iclm_cvtB, dim3((KPAD / 64) * (NM / 64)), dim3(256), 0, stream,
                       prev_state, Bt);
    hipLaunchKernelGGL(iclm_main, dim3(N_ENT), dim3(256), 0, stream,
                       input_all, Bt, ws, out);
}
